// NodeEdgeCrossAttention_22016002359713
// MI455X (gfx1250) — compile-verified, matrix operand order checked by reading, not run on hardware
//
#include <hip/hip_runtime.h>

#define DIM   128
#define HEADS 4
#define DHEAD 32
#define LDS_PITCH 136   // halves per row (128 + 8 pad) -> conflict-free column reads

typedef __attribute__((ext_vector_type(16))) _Float16 v16h;
typedef __attribute__((ext_vector_type(8)))  float    v8f;

// ---------------------------------------------------------------------------
// K0: convert fp32 weight W[k][n] -> fp16 transposed Wt[n][k]
// ---------------------------------------------------------------------------
__global__ void nea_wt_transpose(const float* __restrict__ W, _Float16* __restrict__ Wt) {
    int idx = blockIdx.x * blockDim.x + threadIdx.x;      // over 128*128 (exact grid)
    int n = idx >> 7, k = idx & 127;
    Wt[n * DIM + k] = (_Float16)W[k * DIM + n];
}

// ---------------------------------------------------------------------------
// K1: Y[M x 128] = X[M x 128] @ W + bias, via v_wmma_f32_16x16x32_f16.
// Wt is fp16, row-major [n][k] (i.e. W transposed). 8 waves/block, each wave
// owns a 16-row M tile; Wt staged in LDS (padded pitch). Full tiles take a
// wave-uniform unpredicated path; only the single ragged tile predicates.
// ---------------------------------------------------------------------------
__global__ __launch_bounds__(256)
void nea_gemm128_bias_wmma(const float* __restrict__ X, const _Float16* __restrict__ Wt,
                           const float* __restrict__ bias, float* __restrict__ Y, int M) {
    __shared__ _Float16 sW[DIM * LDS_PITCH];

    // stage weights: 128 rows x 64 dwords -> padded 68-dword pitch
    {
        const uint32_t* src = (const uint32_t*)Wt;
        uint32_t*       dst = (uint32_t*)sW;
        for (int i = threadIdx.x; i < DIM * 64; i += 256) {
            int r = i >> 6, c = i & 63;
            dst[r * (LDS_PITCH / 2) + c] = src[i];
        }
    }
    __syncthreads();

    const int  lane = threadIdx.x & 31;
    const int  wave = threadIdx.x >> 5;
    const int  row0 = blockIdx.x * 128 + wave * 16;
    const int  hi   = (lane & 16) ? 8 : 0;   // K sub-offset for lanes 16..31, also D row offset
    const bool full = (row0 + 16 <= M);      // wave-uniform: tile fully in range
    int arow = row0 + (lane & 15);
    if (!full && arow > M - 1) arow = M - 1; // clamp reads only in the ragged tile

    // Load the 4 A fragments (K = 0,32,64,96), fp32 -> fp16 on the fly.
    // Per-lane ISA layout: elems 0..7 = K k0..k0+7, elems 8..15 = K k0+16..k0+23.
    v16h afrag[4];
    const float* xr = X + (size_t)arow * DIM;
#pragma unroll
    for (int ks = 0; ks < 4; ++ks) {
        int k0 = ks * 32 + hi;
        float4 f0 = *(const float4*)(xr + k0);
        float4 f1 = *(const float4*)(xr + k0 + 4);
        float4 g0 = *(const float4*)(xr + k0 + 16);
        float4 g1 = *(const float4*)(xr + k0 + 20);
        v16h a;
        a[0]  = (_Float16)f0.x; a[1]  = (_Float16)f0.y; a[2]  = (_Float16)f0.z; a[3]  = (_Float16)f0.w;
        a[4]  = (_Float16)f1.x; a[5]  = (_Float16)f1.y; a[6]  = (_Float16)f1.z; a[7]  = (_Float16)f1.w;
        a[8]  = (_Float16)g0.x; a[9]  = (_Float16)g0.y; a[10] = (_Float16)g0.z; a[11] = (_Float16)g0.w;
        a[12] = (_Float16)g1.x; a[13] = (_Float16)g1.y; a[14] = (_Float16)g1.z; a[15] = (_Float16)g1.w;
        afrag[ks] = a;
    }

    const int ncl   = lane & 15;
    const int rbase = row0 + hi;
#pragma unroll
    for (int nt = 0; nt < 8; ++nt) {
        int n = nt * 16 + ncl;
        v8f acc = {};
#pragma unroll
        for (int ks = 0; ks < 4; ++ks) {
            int k0 = ks * 32 + hi;
            const _Float16* bp = sW + n * LDS_PITCH + k0;   // column n of W, contiguous in K
            v16h b;
#pragma unroll
            for (int i = 0; i < 8; ++i) { b[i] = bp[i]; b[8 + i] = bp[16 + i]; }
            acc = __builtin_amdgcn_wmma_f32_16x16x32_f16(
                false, afrag[ks], false, b, (short)0, acc, false, false);
        }
        float  bv = bias[n];
        float* yp = Y + (size_t)rbase * DIM + n;
        if (full) {
#pragma unroll
            for (int i = 0; i < 8; ++i) yp[(size_t)i * DIM] = acc[i] + bv;
        } else {
#pragma unroll
            for (int i = 0; i < 8; ++i) {
                if (rbase + i < M) yp[(size_t)i * DIM] = acc[i] + bv;
            }
        }
    }
}

// ---------------------------------------------------------------------------
// Ordered-uint encoding of float so atomicMax(uint) == float max; 0 == -inf.
// ---------------------------------------------------------------------------
__device__ __forceinline__ unsigned nea_f2ord(float f) {
    unsigned u = __float_as_uint(f);
    return (u & 0x80000000u) ? ~u : (u | 0x80000000u);
}
__device__ __forceinline__ float nea_ord2f(unsigned u) {
    return __uint_as_float((u & 0x80000000u) ? (u ^ 0x80000000u) : ~u);
}

// ---------------------------------------------------------------------------
// K2: per (edge, head) score = dot(q[dst], k[e]) * scale; segment max via atomics
// ---------------------------------------------------------------------------
__global__ void nea_score(const float* __restrict__ q, const float* __restrict__ k,
                          const int* __restrict__ dst, float* __restrict__ score,
                          unsigned* __restrict__ segmax, int E) {
    int t = blockIdx.x * blockDim.x + threadIdx.x;
    if (t >= E * HEADS) return;
    int e = t >> 2, h = t & 3;
    int d = dst[e];
    const float4* kp = (const float4*)(k + (size_t)e * DIM + h * DHEAD);
    const float4* qp = (const float4*)(q + (size_t)d * DIM + h * DHEAD);
    float s = 0.f;
#pragma unroll
    for (int i = 0; i < 8; ++i) {
        float4 kv = kp[i], qv = qp[i];
        s += kv.x * qv.x + kv.y * qv.y + kv.z * qv.z + kv.w * qv.w;
    }
    s *= 0.17677669529663687f;   // 1/sqrt(32)
    score[t] = s;
    atomicMax(&segmax[d * HEADS + h], nea_f2ord(s));
}

// ---------------------------------------------------------------------------
// K3: ex = exp(score - segmax[dst]); denom[dst] += ex (segment sum)
// ---------------------------------------------------------------------------
__global__ void nea_expsum(const int* __restrict__ dst, const unsigned* __restrict__ segmax,
                           float* __restrict__ score, float* __restrict__ denom, int E) {
    int t = blockIdx.x * blockDim.x + threadIdx.x;
    if (t >= E * HEADS) return;
    int e = t >> 2, h = t & 3;
    int d = dst[e];
    float m  = nea_ord2f(segmax[d * HEADS + h]);
    float ex = expf(score[t] - m);
    score[t] = ex;                               // reuse buffer: now holds exp
    atomicAdd(&denom[d * HEADS + h], ex);
}

// ---------------------------------------------------------------------------
// K4: agg[dst] += v[e] * (ex / denom[dst]) — one thread per float4 chunk
// ---------------------------------------------------------------------------
__global__ void nea_aggregate(const int* __restrict__ dst, const float* __restrict__ ex,
                              const float* __restrict__ denom, const float* __restrict__ v,
                              float* __restrict__ agg, int E) {
    int t = blockIdx.x * blockDim.x + threadIdx.x;   // over E*32
    if (t >= E * 32) return;
    int e = t >> 5;
    int j = (t & 31) * 4;        // float offset within the 128-wide row
    int h = j >> 5;              // head = j / DHEAD
    int d = dst[e];
    float a = ex[e * HEADS + h] / denom[d * HEADS + h];
    float4 vv = *(const float4*)(v + (size_t)e * DIM + j);
    float* ap = agg + (size_t)d * DIM + j;
    atomicAdd(ap + 0, vv.x * a);
    atomicAdd(ap + 1, vv.y * a);
    atomicAdd(ap + 2, vv.z * a);
    atomicAdd(ap + 3, vv.w * a);
}

// ---------------------------------------------------------------------------
extern "C" void kernel_launch(void* const* d_in, const int* in_sizes, int n_in,
                              void* d_out, int out_size, void* d_ws, size_t ws_size,
                              hipStream_t stream) {
    const float* q_nodes = (const float*)d_in[0];
    const float* k_edges = (const float*)d_in[1];
    const float* v_edges = (const float*)d_in[2];
    const float* Wq = (const float*)d_in[3];
    const float* bq = (const float*)d_in[4];
    const float* Wk = (const float*)d_in[5];
    const float* bk = (const float*)d_in[6];
    const float* Wv = (const float*)d_in[7];
    const float* bv = (const float*)d_in[8];
    const float* Wo = (const float*)d_in[9];
    const float* bo = (const float*)d_in[10];
    const int*   edge_index = (const int*)d_in[11];   // [2,E]; row 0 = dst
    const int N = in_sizes[0] / DIM;
    const int E = in_sizes[1] / DIM;
    float* out = (float*)d_out;
    const int* dst = edge_index;                      // first E entries

    // ---- carve workspace (256B aligned slices) ----
    char* wp = (char*)d_ws;
    auto carve = [&](size_t bytes) -> char* {
        char* p = wp;
        wp += (bytes + 255) & ~(size_t)255;
        return p;
    };
    _Float16* wtq = (_Float16*)carve((size_t)DIM * DIM * sizeof(_Float16));
    _Float16* wtk = (_Float16*)carve((size_t)DIM * DIM * sizeof(_Float16));
    _Float16* wtv = (_Float16*)carve((size_t)DIM * DIM * sizeof(_Float16));
    _Float16* wto = (_Float16*)carve((size_t)DIM * DIM * sizeof(_Float16));
    float*    qb  = (float*)carve((size_t)N * DIM * sizeof(float));
    float*    kb2 = (float*)carve((size_t)E * DIM * sizeof(float));
    float*    vb2 = (float*)carve((size_t)E * DIM * sizeof(float));
    float*    sc  = (float*)carve((size_t)E * HEADS * sizeof(float));
    unsigned* smx = (unsigned*)carve((size_t)N * HEADS * sizeof(unsigned));
    float*    den = (float*)carve((size_t)N * HEADS * sizeof(float));
    float*    agg = (float*)carve((size_t)N * DIM * sizeof(float));

    // ---- zero accumulators every call (graph-capture-safe) ----
    hipMemsetAsync(smx, 0, (size_t)N * HEADS * sizeof(unsigned), stream); // 0 == -inf (ordered)
    hipMemsetAsync(den, 0, (size_t)N * HEADS * sizeof(float), stream);
    hipMemsetAsync(agg, 0, (size_t)N * DIM * sizeof(float), stream);

    // ---- weight transpose + fp16 convert ----
    nea_wt_transpose<<<(DIM * DIM) / 256, 256, 0, stream>>>(Wq, wtq);
    nea_wt_transpose<<<(DIM * DIM) / 256, 256, 0, stream>>>(Wk, wtk);
    nea_wt_transpose<<<(DIM * DIM) / 256, 256, 0, stream>>>(Wv, wtv);
    nea_wt_transpose<<<(DIM * DIM) / 256, 256, 0, stream>>>(Wo, wto);

    // ---- projections (WMMA GEMMs) ----
    nea_gemm128_bias_wmma<<<(N + 127) / 128, 256, 0, stream>>>(q_nodes, wtq, bq, qb, N);
    nea_gemm128_bias_wmma<<<(E + 127) / 128, 256, 0, stream>>>(k_edges, wtk, bk, kb2, E);
    nea_gemm128_bias_wmma<<<(E + 127) / 128, 256, 0, stream>>>(v_edges, wtv, bv, vb2, E);

    // ---- segment softmax + aggregation ----
    int ehThreads = E * HEADS;
    nea_score  <<<(ehThreads + 255) / 256, 256, 0, stream>>>(qb, kb2, dst, sc, smx, E);
    nea_expsum <<<(ehThreads + 255) / 256, 256, 0, stream>>>(dst, smx, sc, den, E);
    nea_aggregate<<<((size_t)E * 32 + 255) / 256, 256, 0, stream>>>(dst, sc, den, vb2, agg, E);

    // ---- output projection straight into d_out ----
    nea_gemm128_bias_wmma<<<(N + 127) / 128, 256, 0, stream>>>(agg, wto, bo, out, N);
}
